// CausalMultiHeadAttention_18296560681482
// MI455X (gfx1250) — matrix operand order repaired and checked by reading, not compiled, not run on hardware
//
#include <hip/hip_runtime.h>
#include <hip/hip_bf16.h>

typedef __attribute__((ext_vector_type(16))) _Float16 v16h;
typedef __attribute__((ext_vector_type(8)))  _Float16 v8h;
typedef __attribute__((ext_vector_type(8)))  float    v8f;

#define BATCH  2
#define SEQ    2048
#define DIMM   1024
#define NHEADS 16
#define DHEAD  64
#define ROWS   (BATCH * SEQ)   // 4096

union F8  { v8f  v; float    f[8];  };
union H16 { v16h v; v8h h[2]; _Float16 e[16]; };

__device__ __forceinline__ v8f wmma_f16(v16h a, v16h b, v8f c) {
  return __builtin_amdgcn_wmma_f32_16x16x32_f16(false, a, false, b, (short)0, c, false, false);
}

// A fragment: 16x32 (MxK) f16. Source row-major, leading dim `ld` (halves).
// ISA layout: lane m = l&15, half = l>>4; a[0..7] = A[m][half*8 + 0..7],
// a[8..15] = A[m][16 + half*8 + 0..7].
__device__ __forceinline__ v16h load_frag_a(const _Float16* src, int ld) {
  int lane = threadIdx.x & 31;
  int m = lane & 15, hf = lane >> 4;
  const _Float16* p = src + (size_t)m * ld + hf * 8;
  H16 u;
  u.h[0] = *(const v8h*)(p);
  u.h[1] = *(const v8h*)(p + 16);
  return u.v;
}

// B fragment: 32x16 (KxN) where B[k][n] = W[n][k]; W row-major, ld halves.
// ISA layout: lane n = l&15, khalf = l>>4; b[0..7] = W[n][khalf*8 + 0..7],
// b[8..15] = W[n][16 + khalf*8 + 0..7].
__device__ __forceinline__ v16h load_frag_b(const _Float16* W, int ld) {
  int lane = threadIdx.x & 31;
  int n = lane & 15, kh = lane >> 4;
  const _Float16* p = W + (size_t)n * ld + kh * 8;
  H16 u;
  u.h[0] = *(const v8h*)(p);
  u.h[1] = *(const v8h*)(p + 16);
  return u.v;
}

// ---------------------------------------------------------------- converts
__global__ __launch_bounds__(256) void f32_to_f16_kernel(const float* __restrict__ in,
                                                         _Float16* __restrict__ out, int n) {
  int i = blockIdx.x * 256 + threadIdx.x;
  if (i < n) out[i] = (_Float16)in[i];
}

// ---------------------------------------------------------------- LayerNorm (f32 in -> f16 out)
__global__ __launch_bounds__(256) void ln_f32_to_f16(const float* __restrict__ x,
                                                     const float* __restrict__ g,
                                                     const float* __restrict__ bb,
                                                     _Float16* __restrict__ out) {
  const int D = DIMM;
  int row = blockIdx.x;
  const float* xr = x + (size_t)row * D;
  float vals[4];
  float s = 0.f, s2 = 0.f;
  for (int i = 0; i < 4; ++i) {
    float v = xr[threadIdx.x + i * 256];
    vals[i] = v; s += v; s2 += v * v;
  }
  for (int off = 16; off; off >>= 1) {
    s  += __shfl_xor(s,  off, 32);
    s2 += __shfl_xor(s2, off, 32);
  }
  __shared__ float rs[8], rs2[8];
  int wid = threadIdx.x >> 5;
  if ((threadIdx.x & 31) == 0) { rs[wid] = s; rs2[wid] = s2; }
  __syncthreads();
  float S = 0.f, S2 = 0.f;
  for (int i = 0; i < 8; ++i) { S += rs[i]; S2 += rs2[i]; }
  float mu   = S * (1.f / D);
  float var  = S2 * (1.f / D) - mu * mu;
  float rstd = rsqrtf(var + 1e-5f);
  for (int i = 0; i < 4; ++i) {
    int c = threadIdx.x + i * 256;
    out[(size_t)row * D + c] = (_Float16)((vals[i] - mu) * rstd * g[c] + bb[c]);
  }
}

// ---------------------------------------------------------------- LayerNorm (f16 in -> f32 out)
__global__ __launch_bounds__(256) void ln_f16_to_f32(const _Float16* __restrict__ x,
                                                     const float* __restrict__ g,
                                                     const float* __restrict__ bb,
                                                     float* __restrict__ out) {
  const int D = DIMM;
  int row = blockIdx.x;
  const _Float16* xr = x + (size_t)row * D;
  float vals[4];
  float s = 0.f, s2 = 0.f;
  for (int i = 0; i < 4; ++i) {
    float v = (float)xr[threadIdx.x + i * 256];
    vals[i] = v; s += v; s2 += v * v;
  }
  for (int off = 16; off; off >>= 1) {
    s  += __shfl_xor(s,  off, 32);
    s2 += __shfl_xor(s2, off, 32);
  }
  __shared__ float rs[8], rs2[8];
  int wid = threadIdx.x >> 5;
  if ((threadIdx.x & 31) == 0) { rs[wid] = s; rs2[wid] = s2; }
  __syncthreads();
  float S = 0.f, S2 = 0.f;
  for (int i = 0; i < 8; ++i) { S += rs[i]; S2 += rs2[i]; }
  float mu   = S * (1.f / D);
  float var  = S2 * (1.f / D) - mu * mu;
  float rstd = rsqrtf(var + 1e-5f);
  for (int i = 0; i < 4; ++i) {
    int c = threadIdx.x + i * 256;
    out[(size_t)row * D + c] = (vals[i] - mu) * rstd * g[c] + bb[c];
  }
}

// ---------------------------------------------------------------- NT GEMM: C[M,N] = A[M,K] * W[N,K]^T
// 256 threads = 8 waves arranged 4(M) x 2(N); block tile 256(M) x 128(N);
// wave tile 64(M) x 64(N): 16 WMMAs per k-step vs 8 fragment loads
// (1.0 16B-load/WMMA) -- matrix pipes stay fed, VMEM issue port relieved.
// ~210 VGPRs/wave, well under the CDNA5 1024-VGPR ceiling.
__global__ __launch_bounds__(256) void gemm_nt_f16(const _Float16* __restrict__ A,
                                                   const _Float16* __restrict__ W,
                                                   _Float16* __restrict__ C,
                                                   int M, int N, int K, float scale) {
  int wave = threadIdx.x >> 5;
  int wm = wave & 3;        // 0..3
  int wn = wave >> 2;       // 0..1
  int m0 = blockIdx.y * 256 + wm * 64;
  int n0 = blockIdx.x * 128 + wn * 64;
  if (m0 >= M || n0 >= N) return;

  v8f zero = {};
  F8 acc[4][4];
#pragma unroll
  for (int mt = 0; mt < 4; ++mt)
#pragma unroll
    for (int nt = 0; nt < 4; ++nt) acc[mt][nt].v = zero;

  for (int k0 = 0; k0 < K; k0 += 32) {
    v16h af[4];
#pragma unroll
    for (int mt = 0; mt < 4; ++mt)
      af[mt] = load_frag_a(A + (size_t)(m0 + mt * 16) * K + k0, K);
    v16h bf[4];
#pragma unroll
    for (int nt = 0; nt < 4; ++nt)
      bf[nt] = load_frag_b(W + (size_t)(n0 + nt * 16) * K + k0, K);
#pragma unroll
    for (int mt = 0; mt < 4; ++mt)
#pragma unroll
      for (int nt = 0; nt < 4; ++nt)
        acc[mt][nt].v = wmma_f16(af[mt], bf[nt], acc[mt][nt].v);
  }

  int lane = threadIdx.x & 31;
  int col = lane & 15, hf = lane >> 4;
#pragma unroll
  for (int mt = 0; mt < 4; ++mt)
#pragma unroll
    for (int nt = 0; nt < 4; ++nt)
#pragma unroll
      for (int v = 0; v < 8; ++v) {
        int m = v + 8 * hf;
        C[(size_t)(m0 + mt * 16 + m) * N + n0 + nt * 16 + col] =
            (_Float16)(acc[mt][nt].f[v] * scale);
      }
}

// ---------------------------------------------------------------- kv reformat: k scaled row-major, v transposed
__global__ __launch_bounds__(256) void reformat_kv(const _Float16* __restrict__ kv,
                                                   _Float16* __restrict__ kout,
                                                   _Float16* __restrict__ vT) {
  int idx = blockIdx.x * 256 + threadIdx.x;           // over ROWS*DHEAD
  if (idx >= ROWS * DHEAD) return;
  int r = idx >> 6, d = idx & 63;                     // r = b*SEQ + n
  int b = r >> 11, n = r & (SEQ - 1);
  const float sqs = 0.3535533905932738f;              // (1/8)^0.5
  kout[(size_t)r * DHEAD + d] = (_Float16)((float)kv[(size_t)r * 128 + d] * sqs);
  vT[((size_t)(b * DHEAD + d)) * SEQ + n] = kv[(size_t)r * 128 + 64 + d];
}

// ---------------------------------------------------------------- causal flash attention, one wave per (b,h,qtile)
__global__ __launch_bounds__(32) void attn_kernel(const _Float16* __restrict__ qf,  // [ROWS, DIMM]
                                                  const _Float16* __restrict__ kf,  // [ROWS, DHEAD] scaled
                                                  const _Float16* __restrict__ vT,  // [B*DHEAD, SEQ]
                                                  _Float16* __restrict__ out) {     // [ROWS, DIMM]
  const int QT = SEQ / 16;                            // 128 query tiles
  int blk = blockIdx.x;                               // b*NHEADS*QT + h*QT + qt
  int qt = blk % QT;
  int h  = (blk / QT) % NHEADS;
  int b  = blk / (QT * NHEADS);

  int lane = threadIdx.x & 31;
  int nn = lane & 15, hf = lane >> 4;

  __shared__ __align__(16) _Float16 Pld[16 * 32];

  const _Float16* Qbase = qf + (size_t)(b * SEQ + qt * 16) * DIMM + h * DHEAD;
  v16h qfrag0 = load_frag_a(Qbase + 0,  DIMM);
  v16h qfrag1 = load_frag_a(Qbase + 32, DIMM);

  v8f zero = {};
  F8 O[4];
  for (int t = 0; t < 4; ++t) O[t].v = zero;
  float mstate[8], lstate[8];
#pragma unroll
  for (int v = 0; v < 8; ++v) { mstate[v] = -1e30f; lstate[v] = 0.f; }

  int jmax = (qt * 16 + 15) >> 5;
  for (int j = 0; j <= jmax; ++j) {
    int kb = j * 32;
    // ---- S = Q * K^T (16 queries x 32 keys)
    F8 S[2];
#pragma unroll
    for (int t = 0; t < 2; ++t) {
      const _Float16* Kb = kf + (size_t)(b * SEQ + kb + t * 16) * DHEAD;
      v16h b0 = load_frag_b(Kb + 0,  DHEAD);
      v16h b1 = load_frag_b(Kb + 32, DHEAD);
      v8f s = zero;
      s = wmma_f16(qfrag0, b0, s);
      s = wmma_f16(qfrag1, b1, s);
      S[t].v = s;
      // causal mask: key index > query index -> -inf
#pragma unroll
      for (int v = 0; v < 8; ++v) {
        int gq = qt * 16 + v + 8 * hf;
        int gk = kb + t * 16 + nn;
        if (gk > gq) S[t].f[v] = -1e30f;
      }
    }
    // ---- row max across the 16 lanes holding each row
    float rmax[8];
#pragma unroll
    for (int v = 0; v < 8; ++v) {
      float r = fmaxf(S[0].f[v], S[1].f[v]);
      for (int m = 1; m < 16; m <<= 1) r = fmaxf(r, __shfl_xor(r, m, 32));
      rmax[v] = r;
    }
    float alpha[8];
#pragma unroll
    for (int v = 0; v < 8; ++v) {
      float mn = fmaxf(mstate[v], rmax[v]);
      alpha[v] = __expf(mstate[v] - mn);
      mstate[v] = mn;
    }
    // ---- P = exp(S - m); row sums
    float rsum[8];
#pragma unroll
    for (int v = 0; v < 8; ++v) {
      S[0].f[v] = __expf(S[0].f[v] - mstate[v]);
      S[1].f[v] = __expf(S[1].f[v] - mstate[v]);
      float r = S[0].f[v] + S[1].f[v];
      for (int m = 1; m < 16; m <<= 1) r += __shfl_xor(r, m, 32);
      rsum[v] = r;
    }
#pragma unroll
    for (int v = 0; v < 8; ++v) lstate[v] = lstate[v] * alpha[v] + rsum[v];
#pragma unroll
    for (int t = 0; t < 4; ++t)
#pragma unroll
      for (int v = 0; v < 8; ++v) O[t].f[v] *= alpha[v];
    // ---- re-layout P (C layout) -> A-fragment via LDS (single wave: DS ops are in-order)
#pragma unroll
    for (int t = 0; t < 2; ++t)
#pragma unroll
      for (int v = 0; v < 8; ++v)
        Pld[(v + 8 * hf) * 32 + t * 16 + nn] = (_Float16)S[t].f[v];
    v16h pfrag = load_frag_a(&Pld[0], 32);
    // ---- O += P * V
#pragma unroll
    for (int nt = 0; nt < 4; ++nt) {
      v16h vfrag = load_frag_b(vT + (size_t)(b * DHEAD + nt * 16) * SEQ + kb, SEQ);
      O[nt].v = wmma_f16(pfrag, vfrag, O[nt].v);
    }
  }
  // ---- normalize and store
#pragma unroll
  for (int nt = 0; nt < 4; ++nt)
#pragma unroll
    for (int v = 0; v < 8; ++v) {
      int m = v + 8 * hf;
      float o = O[nt].f[v] / lstate[v];
      out[(size_t)(b * SEQ + qt * 16 + m) * DIMM + h * DHEAD + nt * 16 + nn] = (_Float16)o;
    }
}

// ---------------------------------------------------------------- launch
extern "C" void kernel_launch(void* const* d_in, const int* in_sizes, int n_in,
                              void* d_out, int out_size, void* d_ws, size_t ws_size,
                              hipStream_t stream) {
  (void)in_sizes; (void)n_in; (void)out_size; (void)ws_size;
  const float* x     = (const float*)d_in[0];
  const float* w_q   = (const float*)d_in[1];
  const float* w_kv  = (const float*)d_in[2];
  const float* w_out = (const float*)d_in[3];
  const float* ln_g  = (const float*)d_in[4];
  const float* ln_b  = (const float*)d_in[5];
  const float* lno_g = (const float*)d_in[6];
  const float* lno_b = (const float*)d_in[7];
  float* out = (float*)d_out;

  size_t off = 0;
  char* base = (char*)d_ws;
  auto carve = [&](size_t bytes) { char* p = base + off; off += (bytes + 255) & ~(size_t)255; return p; };

  _Float16* xn16   = (_Float16*)carve((size_t)ROWS * DIMM * 2);        // 8 MB
  _Float16* wq16   = (_Float16*)carve((size_t)DIMM * DIMM * 2);        // 2 MB
  _Float16* wkv16  = (_Float16*)carve((size_t)128 * DIMM * 2);         // 256 KB
  _Float16* wout16 = (_Float16*)carve((size_t)DIMM * DIMM * 2);        // 2 MB
  _Float16* q16    = (_Float16*)carve((size_t)ROWS * DIMM * 2);        // 8 MB
  _Float16* kv16   = (_Float16*)carve((size_t)ROWS * 128 * 2);         // 1 MB
  _Float16* k16    = (_Float16*)carve((size_t)ROWS * DHEAD * 2);       // 512 KB
  _Float16* vT16   = (_Float16*)carve((size_t)BATCH * DHEAD * SEQ * 2);// 512 KB
  _Float16* ao16   = (_Float16*)carve((size_t)ROWS * DIMM * 2);        // 8 MB
  _Float16* o216   = (_Float16*)carve((size_t)ROWS * DIMM * 2);        // 8 MB

  const float qscale = 0.04419417382415922f;  // (1/8) * (1/8)^0.5

  // 1) LayerNorm + f16 conversion of activations and weights
  ln_f32_to_f16<<<ROWS, 256, 0, stream>>>(x, ln_g, ln_b, xn16);
  f32_to_f16_kernel<<<(DIMM * DIMM + 255) / 256, 256, 0, stream>>>(w_q,   wq16,   DIMM * DIMM);
  f32_to_f16_kernel<<<(128 * DIMM + 255) / 256,  256, 0, stream>>>(w_kv,  wkv16,  128 * DIMM);
  f32_to_f16_kernel<<<(DIMM * DIMM + 255) / 256, 256, 0, stream>>>(w_out, wout16, DIMM * DIMM);

  // 2) q / kv projections (WMMA)
  gemm_nt_f16<<<dim3(DIMM / 128, ROWS / 256), 256, 0, stream>>>(xn16, wq16,  q16,  ROWS, DIMM, DIMM, qscale);
  gemm_nt_f16<<<dim3(1,          ROWS / 256), 256, 0, stream>>>(xn16, wkv16, kv16, ROWS, 128,  DIMM, 1.0f);

  // 3) split/scale k, transpose v
  reformat_kv<<<(ROWS * DHEAD + 255) / 256, 256, 0, stream>>>(kv16, k16, vT16);

  // 4) causal attention (WMMA flash-style)
  attn_kernel<<<BATCH * NHEADS * (SEQ / 16), 32, 0, stream>>>(q16, k16, vT16, ao16);

  // 5) out projection (WMMA)
  gemm_nt_f16<<<dim3(DIMM / 128, ROWS / 256), 256, 0, stream>>>(ao16, wout16, o216, ROWS, DIMM, DIMM, 1.0f);

  // 6) final LayerNorm -> f32 output
  ln_f16_to_f32<<<ROWS, 256, 0, stream>>>(o216, lno_g, lno_b, out);
}
